// GraphClassifier_34067680592557
// MI455X (gfx1250) — matrix operand order repaired and checked by reading, not compiled, not run on hardware
//
#include <hip/hip_runtime.h>

#define D_IN   256
#define D_HID  512
#define D_OUT_ 48
#define KSTEPS 10
#define ALPHA  0.1f
#define LDS_STRIDE 520   // 512 + 8 bf16 pad -> no LDS bank conflicts
#define SCAN_BS 512

typedef __attribute__((ext_vector_type(16))) __bf16 v16bf;
typedef __attribute__((ext_vector_type(8)))  float  v8f;

// ---------------------------------------------------------------------------
// Prepack W1 [256x512] f32 -> bf16 in WMMA B-fragment order.
// packed[((j*(D_IN/32)+kk)*32 + lane)*16 + e] = W1[k*D_HID + n]
//   n = 16*j + (lane&15) ; k = 32*kk + (e<8 ? 8*(lane>>4)+e : 16+8*(lane>>4)+(e-8))   (B 32x16 layout)
__global__ void prepack_w1_kernel(const float* __restrict__ w1,
                                  __bf16* __restrict__ w1p) {
  int idx = blockIdx.x * blockDim.x + threadIdx.x;
  const int total = (D_HID / 16) * (D_IN / 32) * 32 * 16;  // 131072
  if (idx >= total) return;
  int e    = idx & 15;
  int lane = (idx >> 4) & 31;
  int kk   = (idx >> 9) & (D_IN / 32 - 1);
  int j    = idx >> 12;
  int n = 16 * j + (lane & 15);
  int k = 32 * kk + (e < 8 ? 8 * (lane >> 4) + e : 16 + 8 * (lane >> 4) + (e - 8));
  w1p[idx] = (__bf16)w1[k * D_HID + n];
}

__global__ void prepack_w2_kernel(const float* __restrict__ w2,
                                  __bf16* __restrict__ w2p) {
  int idx = blockIdx.x * blockDim.x + threadIdx.x;
  const int total = (D_OUT_ / 16) * (D_HID / 32) * 32 * 16;  // 24576
  if (idx >= total) return;
  int e    = idx & 15;
  int lane = (idx >> 4) & 31;
  int kk   = (idx >> 9) & (D_HID / 32 - 1);
  int j    = idx >> 13;
  int n = 16 * j + (lane & 15);
  int k = 32 * kk + (e < 8 ? 8 * (lane >> 4) + e : 16 + 8 * (lane >> 4) + (e - 8));
  w2p[idx] = (__bf16)w2[k * D_OUT_ + n];
}

// ---------------------------------------------------------------------------
// Fused MLP: one wave per 32-row block (two 16-row A-tiles share each B tile).
__global__ __launch_bounds__(32) void mlp_wmma_kernel(
    const float* __restrict__ x,
    const __bf16* __restrict__ w1p, const float* __restrict__ b1,
    const __bf16* __restrict__ w2p, const float* __restrict__ b2,
    float* __restrict__ h, int n_nodes) {
  __shared__ __bf16 lds_h1[2][16 * LDS_STRIDE];  // 33 KB
  const int lane = threadIdx.x;
  const int half = lane >> 4;
  const int lm   = lane & 15;
  const int row0 = blockIdx.x * 32;

  int ar0 = row0 + lm;      if (ar0 >= n_nodes) ar0 = n_nodes - 1;
  int ar1 = row0 + 16 + lm; if (ar1 >= n_nodes) ar1 = n_nodes - 1;
  const float* __restrict__ xr0 = x + (size_t)ar0 * D_IN;
  const float* __restrict__ xr1 = x + (size_t)ar1 * D_IN;

  // preload + convert all A fragments for both row tiles (kept in VGPRs)
  v16bf a0[D_IN / 32], a1[D_IN / 32];
#pragma unroll
  for (int kk = 0; kk < D_IN / 32; ++kk) {
    const float* p0 = xr0 + 32 * kk + 8 * half;
    const float* p1 = xr1 + 32 * kk + 8 * half;
#pragma unroll
    for (int e2 = 0; e2 < 8; ++e2) {
      a0[kk][e2]     = (__bf16)p0[e2];
      a0[kk][e2 + 8] = (__bf16)p0[16 + e2];
      a1[kk][e2]     = (__bf16)p1[e2];
      a1[kk][e2 + 8] = (__bf16)p1[16 + e2];
    }
  }

  // -------- stage 1: H1 = relu(X@W1 + b1) -> bf16 tiles in LDS --------
  for (int j = 0; j < D_HID / 16; ++j) {
    v8f acc0 = {0.f, 0.f, 0.f, 0.f, 0.f, 0.f, 0.f, 0.f};
    v8f acc1 = {0.f, 0.f, 0.f, 0.f, 0.f, 0.f, 0.f, 0.f};
    const __bf16* bptr = w1p + (size_t)j * (D_IN / 32) * 512;
#pragma unroll
    for (int kk = 0; kk < D_IN / 32; ++kk) {
      v16bf b = *(const v16bf*)(bptr + ((size_t)kk * 32 + lane) * 16);
      acc0 = __builtin_amdgcn_wmma_f32_16x16x32_bf16(false, a0[kk], false, b,
                                                     (short)0, acc0, false, false);
      acc1 = __builtin_amdgcn_wmma_f32_16x16x32_bf16(false, a1[kk], false, b,
                                                     (short)0, acc1, false, false);
    }
    const float bias = b1[16 * j + lm];
#pragma unroll
    for (int r = 0; r < 8; ++r) {   // C layout: n = lane&15, m = r + 8*(lane>>4)
      const int m = r + 8 * half;
      lds_h1[0][m * LDS_STRIDE + 16 * j + lm] = (__bf16)fmaxf(acc0[r] + bias, 0.f);
      lds_h1[1][m * LDS_STRIDE + 16 * j + lm] = (__bf16)fmaxf(acc1[r] + bias, 0.f);
    }
  }
  __syncthreads();

  // -------- stage 2: H = H1@W2 + b2 --------
  for (int j2 = 0; j2 < D_OUT_ / 16; ++j2) {
    v8f acc0 = {0.f, 0.f, 0.f, 0.f, 0.f, 0.f, 0.f, 0.f};
    v8f acc1 = {0.f, 0.f, 0.f, 0.f, 0.f, 0.f, 0.f, 0.f};
    const __bf16* bptr = w2p + (size_t)j2 * (D_HID / 32) * 512;
#pragma unroll 4
    for (int kk = 0; kk < D_HID / 32; ++kk) {
      v16bf b = *(const v16bf*)(bptr + ((size_t)kk * 32 + lane) * 16);
      const __bf16* q0 = &lds_h1[0][lm * LDS_STRIDE + 32 * kk + 8 * half];
      const __bf16* q1 = &lds_h1[1][lm * LDS_STRIDE + 32 * kk + 8 * half];
      v16bf f0, f1;
#pragma unroll
      for (int e2 = 0; e2 < 8; ++e2) {
        f0[e2] = q0[e2]; f0[e2 + 8] = q0[16 + e2];
        f1[e2] = q1[e2]; f1[e2 + 8] = q1[16 + e2];
      }
      acc0 = __builtin_amdgcn_wmma_f32_16x16x32_bf16(false, f0, false, b,
                                                     (short)0, acc0, false, false);
      acc1 = __builtin_amdgcn_wmma_f32_16x16x32_bf16(false, f1, false, b,
                                                     (short)0, acc1, false, false);
    }
    const float bias = b2[16 * j2 + lm];
#pragma unroll
    for (int r = 0; r < 8; ++r) {
      const int m = r + 8 * half;
      int g0 = row0 + m, g1 = row0 + 16 + m;
      if (g0 < n_nodes) h[(size_t)g0 * D_OUT_ + 16 * j2 + lm] = acc0[r] + bias;
      if (g1 < n_nodes) h[(size_t)g1 * D_OUT_ + 16 * j2 + lm] = acc1[r] + bias;
    }
  }
}

// ---------------------------------------------------------------------------
// CSR build (once) + gather-based APPNP (no atomics in the 10 rounds)
__global__ void zero_cnt_kernel(int* cnt, int n) {
  int i = blockIdx.x * blockDim.x + threadIdx.x;
  if (i < n) cnt[i] = 0;
}

__global__ void count_kernel(const long long* __restrict__ ei, int* __restrict__ cnt,
                             int E) {
  int e = blockIdx.x * blockDim.x + threadIdx.x;
  if (e < E) atomicAdd(&cnt[(int)ei[(size_t)E + e]], 1);
}

__global__ void dinv_kernel(const int* __restrict__ cnt, float* __restrict__ dinv,
                            int n) {
  int i = blockIdx.x * blockDim.x + threadIdx.x;
  if (i < n) dinv[i] = rsqrtf((float)cnt[i] + 1.0f);  // +1 self-loop
}

__global__ void scan1_kernel(const int* __restrict__ cnt, int* __restrict__ excl,
                             int* __restrict__ bsum, int n) {
  __shared__ int sh[SCAN_BS];
  int tid = threadIdx.x;
  int i = blockIdx.x * SCAN_BS + tid;
  int v = (i < n) ? cnt[i] : 0;
  sh[tid] = v;
  __syncthreads();
  for (int off = 1; off < SCAN_BS; off <<= 1) {
    int t = (tid >= off) ? sh[tid - off] : 0;
    __syncthreads();
    sh[tid] += t;
    __syncthreads();
  }
  if (i < n) excl[i] = sh[tid] - v;
  if (tid == SCAN_BS - 1) bsum[blockIdx.x] = sh[tid];
}

__global__ void scan2_kernel(int* __restrict__ bsum, int nb) {  // nb <= SCAN_BS
  __shared__ int sh[SCAN_BS];
  int tid = threadIdx.x;
  int v = (tid < nb) ? bsum[tid] : 0;
  sh[tid] = v;
  __syncthreads();
  for (int off = 1; off < SCAN_BS; off <<= 1) {
    int t = (tid >= off) ? sh[tid - off] : 0;
    __syncthreads();
    sh[tid] += t;
    __syncthreads();
  }
  if (tid < nb) bsum[tid] = sh[tid] - v;  // exclusive block offsets
}

__global__ void scan3_kernel(int* __restrict__ row_start, int* __restrict__ row_next,
                             const int* __restrict__ bsum, int n) {
  int i = blockIdx.x * blockDim.x + threadIdx.x;
  if (i < n) {
    int s = row_start[i] + bsum[i / SCAN_BS];
    row_start[i] = s;
    row_next[i]  = s;
  }
}

__global__ void csr_fill_kernel(const long long* __restrict__ ei,
                                const float* __restrict__ dinv,
                                int* __restrict__ row_next,
                                int* __restrict__ csr_src,
                                float* __restrict__ csr_w, int E) {
  int e = blockIdx.x * blockDim.x + threadIdx.x;
  if (e >= E) return;
  int s = (int)ei[e];
  int d = (int)ei[(size_t)E + e];
  int p = atomicAdd(&row_next[d], 1);
  csr_src[p] = s;
  csr_w[p]   = (1.0f - ALPHA) * dinv[s] * dinv[d];
}

// one full APPNP round, gather form:
// next[i,c] = alpha*h + 0.9*dinv_i^2*prev[i,c] + sum_in-edges w_e * prev[s,c]
__global__ void appnp_round_kernel(const float* __restrict__ h,
                                   const float* __restrict__ prev,
                                   const float* __restrict__ dinv,
                                   const int* __restrict__ row_start,
                                   const int* __restrict__ cnt,
                                   const int* __restrict__ csr_src,
                                   const float* __restrict__ csr_w,
                                   float* __restrict__ next, int n) {
  int node = blockIdx.x * blockDim.y + threadIdx.y;
  if (node >= n) return;
  int c = threadIdx.x;  // 0..47
  float di = dinv[node];
  size_t base = (size_t)node * D_OUT_ + c;
  float acc = ALPHA * h[base] + (1.0f - ALPHA) * di * di * prev[base];
  int p0 = row_start[node];
  int p1 = p0 + cnt[node];
  for (int p = p0; p < p1; ++p) {
    int s   = csr_src[p];   // broadcast across the 48 channel lanes (L0 hit)
    float w = csr_w[p];
    acc += w * prev[(size_t)s * D_OUT_ + c];
  }
  next[base] = acc;
}

// ---------------------------------------------------------------------------
extern "C" void kernel_launch(void* const* d_in, const int* in_sizes, int n_in,
                              void* d_out, int out_size, void* d_ws, size_t ws_size,
                              hipStream_t stream) {
  const float*     x  = (const float*)d_in[0];
  const long long* ei = (const long long*)d_in[1];
  const float*     W1 = (const float*)d_in[2];
  const float*     b1 = (const float*)d_in[3];
  const float*     W2 = (const float*)d_in[4];
  const float*     b2 = (const float*)d_in[5];
  float* out = (float*)d_out;

  const int N = in_sizes[0] / D_IN;  // 100000
  const int E = in_sizes[1] / 2;     // 1600000
  const int NB = (N + SCAN_BS - 1) / SCAN_BS;  // 196 (< SCAN_BS)

  char* ws = (char*)d_ws;
  size_t off = 0;
  auto alloc = [&](size_t bytes) -> void* {
    void* p = ws + off;
    off = (off + bytes + 255) & ~(size_t)255;
    return p;
  };
  float*  h         = (float*)alloc((size_t)N * D_OUT_ * sizeof(float));
  float*  buf0      = (float*)alloc((size_t)N * D_OUT_ * sizeof(float));
  float*  dinv      = (float*)alloc((size_t)N * sizeof(float));
  int*    cnt       = (int*)alloc((size_t)N * sizeof(int));
  int*    row_start = (int*)alloc((size_t)N * sizeof(int));
  int*    row_next  = (int*)alloc((size_t)N * sizeof(int));
  int*    bsum      = (int*)alloc((size_t)SCAN_BS * sizeof(int));
  int*    csr_src   = (int*)alloc((size_t)E * sizeof(int));
  float*  csr_w     = (float*)alloc((size_t)E * sizeof(float));
  __bf16* w1p       = (__bf16*)alloc((size_t)(D_HID / 16) * (D_IN / 32) * 512 * 2);
  __bf16* w2p       = (__bf16*)alloc((size_t)(D_OUT_ / 16) * (D_HID / 32) * 512 * 2);

  // weight prepack (bf16 WMMA B-fragment order)
  prepack_w1_kernel<<<(131072 + 255) / 256, 256, 0, stream>>>(W1, w1p);
  prepack_w2_kernel<<<(24576 + 255) / 256, 256, 0, stream>>>(W2, w2p);

  // degree count -> dinv -> exclusive scan -> CSR fill
  zero_cnt_kernel<<<(N + 255) / 256, 256, 0, stream>>>(cnt, N);
  count_kernel<<<(E + 255) / 256, 256, 0, stream>>>(ei, cnt, E);
  dinv_kernel<<<(N + 255) / 256, 256, 0, stream>>>(cnt, dinv, N);
  scan1_kernel<<<NB, SCAN_BS, 0, stream>>>(cnt, row_start, bsum, N);
  scan2_kernel<<<1, SCAN_BS, 0, stream>>>(bsum, NB);
  scan3_kernel<<<(N + 255) / 256, 256, 0, stream>>>(row_start, row_next, bsum, N);
  csr_fill_kernel<<<(E + 255) / 256, 256, 0, stream>>>(ei, dinv, row_next, csr_src,
                                                       csr_w, E);

  // fused MLP encoder (bf16 WMMA, 32 rows/wave)
  mlp_wmma_kernel<<<(N + 31) / 32, 32, 0, stream>>>(x, w1p, b1, w2p, b2, h, N);

  // APPNP: 10 gather rounds, ping-pong buf0 <-> d_out so round 9 lands in d_out
  const float* cur = h;
  for (int step = 0; step < KSTEPS; ++step) {
    float* nxt = (step & 1) ? out : buf0;
    dim3 blk(48, 4);
    appnp_round_kernel<<<(N + 3) / 4, blk, 0, stream>>>(
        h, cur, dinv, row_start, cnt, csr_src, csr_w, nxt, N);
    cur = nxt;
  }
}
